// SelfAttentionAttentionPooling_7782480740635
// MI455X (gfx1250) — matrix operand order repaired and checked by reading, not compiled, not run on hardware
//
#include <hip/hip_runtime.h>
#include <hip/hip_bf16.h>

#define B_ 16
#define S_ 2048
#define D_ 512
#define MQ 16          // query rows per block
#define NK 64          // keys per tile
#define NIT (S_ / NK)  // 32 key-tile iterations
#define THREADS 256

typedef __attribute__((ext_vector_type(16))) __bf16       v16bf;
typedef __attribute__((ext_vector_type(8)))  float        v8f;
typedef __attribute__((ext_vector_type(8)))  unsigned int v8u;
typedef __attribute__((ext_vector_type(4)))  unsigned int u32x4;
typedef __attribute__((ext_vector_type(4)))  float        f32x4;
typedef __attribute__((ext_vector_type(2)))  float        f32x2;

// fp32 -> bf16 with round-to-nearest-even
__device__ __forceinline__ unsigned short f2bf(float f) {
  unsigned int u = __float_as_uint(f);
  u += 0x7FFFu + ((u >> 16) & 1u);
  return (unsigned short)(u >> 16);
}

// WMMA A-operand (16x32 bf16, M x K) from row-major LDS.
// Per ISA: lane m=L%16, khalf=L/16; VGPR0-3 hold K = khalf*8 + [0..8),
// VGPR4-7 hold K = 16 + khalf*8 + [0..8). Two contiguous 16B LDS loads.
__device__ __forceinline__ v16bf loadA16(const unsigned short* p, int stride,
                                         int row, int k0, int khalf) {
  const unsigned short* base = p + row * stride + k0 + khalf * 8;
  u32x4 lo = *(const u32x4*)(base);
  u32x4 hi = *(const u32x4*)(base + 16);
  v8u u = {lo.x, lo.y, lo.z, lo.w, hi.x, hi.y, hi.z, hi.w};
  return __builtin_bit_cast(v16bf, u);
}

// WMMA B-operand (32x16 bf16, K x N) read as B^T rows from row-major LDS.
// Per ISA B layout: lane n=L%16, khalf=L/16; VGPR0-3 hold K = khalf*8 + [0..8),
// VGPR4-7 hold K = 16 + khalf*8 + [0..8). Two contiguous 16B LDS loads.
__device__ __forceinline__ v16bf loadB16(const unsigned short* p, int stride,
                                         int row, int k0, int khalf) {
  const unsigned short* base = p + row * stride + k0 + khalf * 8;
  u32x4 lo = *(const u32x4*)(base);
  u32x4 hi = *(const u32x4*)(base + 16);
  v8u u = {lo.x, lo.y, lo.z, lo.w, hi.x, hi.y, hi.z, hi.w};
  return __builtin_bit_cast(v16bf, u);
}

__device__ __forceinline__ v8f wmma_bf16(v16bf a, v16bf b, v8f c) {
  return __builtin_amdgcn_wmma_f32_16x16x32_bf16(false, a, false, b,
                                                 (short)0, c, false, false);
}

// ---------------------------------------------------------------------------
// Kernel 1: flash attention. One block = 16 query rows of one batch.
// Writes y = softmax(x x^T) x  ([B,S,D] fp32) to workspace.
// ---------------------------------------------------------------------------
__global__ void __launch_bounds__(THREADS)
attn_fwd_kernel(const float* __restrict__ x, float* __restrict__ Y) {
  extern __shared__ char smem[];
  unsigned short* tileR = (unsigned short*)smem;                 // [NK][D_] row-major (K for QK^T)
  unsigned short* tileC = (unsigned short*)(smem + NK * D_ * 2); // [D_][NK] col-major (V for PV)
  char* misc = smem + 2 * NK * D_ * 2;                           // 16 KB multiplex region
  unsigned short* qbuf = (unsigned short*)misc;                  // [MQ][D_] (pre-loop staging only)
  float* spart = (float*)misc;                                   // [4][16][16] QK partials (4 KB)
  unsigned short* pbuf = (unsigned short*)(misc + 4096);         // [16][NK] bf16 P tile (2 KB)
  float* stats = (float*)(misc + 8192);
  float* m_lds = stats;                                          // [16] running row max
  float* l_lds = stats + 16;                                     // [16] running row sum
  float* scale_lds = stats + 32;                                 // [16] exp(m_old - m_new)
  float* rowsum = stats + 48;                                    // [16] tile row sums
  float* wmax = stats + 64;                                      // [4][16] per-keyblock row max

  const int tid = threadIdx.x;
  const int w = tid >> 5;         // wave 0..7
  const int lane = tid & 31;
  const int n16 = lane & 15;
  const int khalf = lane >> 4;
  const int wblk = w & 3;         // 16-key column block owned for QK
  const int dOff = (w & 4) ? 256 : 0;  // K-dim split for QK

  const int b = blockIdx.x >> 7;            // 128 blocks per batch
  const int s0 = (blockIdx.x & 127) * MQ;
  const float* xb = x + ((size_t)b * S_) * D_;

  const v8f vzero = {0.f, 0.f, 0.f, 0.f, 0.f, 0.f, 0.f, 0.f};

  // ---- stage Q tile (16x512) as bf16 ----
  for (int i = 0; i < (MQ * D_) / (THREADS * 4); ++i) {  // 8 iters
    int e = (tid + THREADS * i) * 4;
    int r = e >> 9, d = e & (D_ - 1);
    f32x4 f = *(const f32x4*)(xb + (size_t)(s0 + r) * D_ + d);
    unsigned short* q = qbuf + r * D_ + d;
    q[0] = f2bf(f.x); q[1] = f2bf(f.y); q[2] = f2bf(f.z); q[3] = f2bf(f.w);
  }
  // prefetch the first key tile while Q staging settles
  {
    const float* nxt = xb;  // keys [0, NK)
#pragma unroll
    for (int j = 0; j < 4; ++j)
      __builtin_prefetch(nxt + (size_t)(tid + THREADS * j) * 32, 0, 1);
  }
  __syncthreads();
  // Q A-operands stay in registers for the whole K loop (invariant).
  v16bf qreg[8];
#pragma unroll
  for (int c = 0; c < 8; ++c)
    qreg[c] = loadA16(qbuf, D_, n16, dOff + 32 * c, khalf);
  __syncthreads();  // qbuf region is reused below
  if (tid < 16) { m_lds[tid] = -3.0e38f; l_lds[tid] = 0.f; }

  v8f acc[4] = {vzero, vzero, vzero, vzero};  // O slice: 16 rows x 64 d-cols

  for (int it = 0; it < NIT; ++it) {
    const int k0 = it * NK;
    __syncthreads();  // all reads of previous tiles complete
    // ---- load K/V tile (64x512), dual-layout bf16 ----
    for (int i = 0; i < (NK * D_) / (THREADS * 4); ++i) {  // 32 iters
      int e = (tid + THREADS * i) * 4;
      int key = e >> 9, d = e & (D_ - 1);
      f32x4 f = *(const f32x4*)(xb + (size_t)(k0 + key) * D_ + d);
      unsigned short h0 = f2bf(f.x), h1 = f2bf(f.y);
      unsigned short h2 = f2bf(f.z), h3 = f2bf(f.w);
      unsigned short* rp = tileR + key * D_ + d;
      rp[0] = h0; rp[1] = h1; rp[2] = h2; rp[3] = h3;
      tileC[(d + 0) * NK + key] = h0; tileC[(d + 1) * NK + key] = h1;
      tileC[(d + 2) * NK + key] = h2; tileC[(d + 3) * NK + key] = h3;
    }
    // ---- prefetch next key tile (1024 cachelines / 256 threads = 4 each),
    //      lowers to global_prefetch_b8; latency hidden behind 16 WMMAs ----
    if (it + 1 < NIT) {
      const float* nxt = xb + (size_t)(k0 + NK) * D_;
#pragma unroll
      for (int j = 0; j < 4; ++j)
        __builtin_prefetch(nxt + (size_t)(tid + THREADS * j) * 32, 0, 1);
    }
    __syncthreads();

    // ---- S = Q K^T, each wave: 16x16 block, half the K-dim ----
    v8f s = vzero;
#pragma unroll
    for (int c = 0; c < 8; ++c) {
      v16bf bm = loadB16(tileR, D_, wblk * 16 + n16, dOff + 32 * c, khalf);
      s = wmma_bf16(qreg[c], bm, s);
    }
    if (w >= 4) {
#pragma unroll
      for (int r = 0; r < 8; ++r)
        spart[(wblk * 16 + r + 8 * khalf) * 16 + n16] = s[r];
    }
    __syncthreads();
    if (w < 4) {
#pragma unroll
      for (int r = 0; r < 8; ++r) {
        float v = s[r] + spart[(wblk * 16 + r + 8 * khalf) * 16 + n16];
        s[r] = v;
        for (int msk = 1; msk < 16; msk <<= 1)
          v = fmaxf(v, __shfl_xor(v, msk, 32));
        if (n16 == 0) wmax[wblk * 16 + r + 8 * khalf] = v;
      }
    }
    __syncthreads();
    if (tid < 16) {
      float mo = m_lds[tid];
      float mn = mo;
#pragma unroll
      for (int blk = 0; blk < 4; ++blk) mn = fmaxf(mn, wmax[blk * 16 + tid]);
      m_lds[tid] = mn;
      scale_lds[tid] = __expf(mo - mn);
      rowsum[tid] = 0.f;
    }
    __syncthreads();
    // ---- rescale running accumulators (all waves) ----
    {
      float sc[8];
#pragma unroll
      for (int r = 0; r < 8; ++r) sc[r] = scale_lds[r + 8 * khalf];
#pragma unroll
      for (int nt = 0; nt < 4; ++nt)
#pragma unroll
        for (int r = 0; r < 8; ++r) acc[nt][r] *= sc[r];
    }
    // ---- P = exp(S - m), row sums, bf16 P tile ----
    if (w < 4) {
#pragma unroll
      for (int r = 0; r < 8; ++r) {
        float p = __expf(s[r] - m_lds[r + 8 * khalf]);
        pbuf[(r + 8 * khalf) * NK + wblk * 16 + n16] = f2bf(p);
        float t = p;
        for (int msk = 1; msk < 16; msk <<= 1) t += __shfl_xor(t, msk, 32);
        if (n16 == 0) atomicAdd(&rowsum[r + 8 * khalf], t);
      }
    }
    __syncthreads();
    if (tid < 16) l_lds[tid] = l_lds[tid] * scale_lds[tid] + rowsum[tid];
    // ---- O += P V (all waves; wave owns d-cols [64w, 64w+64)) ----
#pragma unroll
    for (int kc = 0; kc < 2; ++kc) {
      v16bf pa = loadA16(pbuf, NK, n16, kc * 32, khalf);
#pragma unroll
      for (int nt = 0; nt < 4; ++nt) {
        v16bf vb = loadB16(tileC, NK, w * 64 + nt * 16 + n16, kc * 32, khalf);
        acc[nt] = wmma_bf16(pa, vb, acc[nt]);
      }
    }
  }
  __syncthreads();

  // ---- epilogue: y = O / l ----
  float linv[8];
#pragma unroll
  for (int r = 0; r < 8; ++r) linv[r] = 1.f / l_lds[r + 8 * khalf];
#pragma unroll
  for (int nt = 0; nt < 4; ++nt) {
#pragma unroll
    for (int r = 0; r < 8; ++r) {
      int row = s0 + r + 8 * khalf;
      int d = w * 64 + nt * 16 + n16;
      Y[((size_t)(b * S_) + row) * D_ + d] = acc[nt][r] * linv[r];
    }
  }
}

// ---------------------------------------------------------------------------
// Kernel 2: attention pooling. One block per batch.
// scores = softmax(y @ query); out = scores^T @ y.
// ---------------------------------------------------------------------------
__global__ void __launch_bounds__(THREADS)
pool_kernel(const float* __restrict__ Y, const float* __restrict__ query,
            float* __restrict__ out_emb, float* __restrict__ out_scores) {
  __shared__ float p_lds[S_];
  __shared__ float red[THREADS];
  __shared__ __align__(16) float qs[D_];
  const int tid = threadIdx.x;
  const int b = blockIdx.x;
  const int w = tid >> 5, lane = tid & 31;
  const float* Yb = Y + (size_t)b * S_ * D_;

  qs[tid] = query[tid];
  qs[tid + 256] = query[tid + 256];
  __syncthreads();

  // phase A: t[s] = y[s,:] . q   (one row per wave, coalesced)
  for (int i = 0; i < S_ / 8; ++i) {
    int s = w + 8 * i;
    const float* yr = Yb + (size_t)s * D_;
    int d0 = lane * 4;
    float sum = 0.f;
#pragma unroll
    for (int j = 0; j < 4; ++j) {
      f32x4 yv = *(const f32x4*)(yr + d0 + 128 * j);
      f32x4 qv = *(const f32x4*)(qs + d0 + 128 * j);
      sum += yv.x * qv.x + yv.y * qv.y + yv.z * qv.z + yv.w * qv.w;
    }
    for (int msk = 1; msk < 32; msk <<= 1) sum += __shfl_xor(sum, msk, 32);
    if (lane == 0) p_lds[s] = sum;
  }
  __syncthreads();

  // block max over 2048
  float mloc = -3.0e38f;
#pragma unroll
  for (int k = 0; k < 8; ++k) mloc = fmaxf(mloc, p_lds[tid + 256 * k]);
  red[tid] = mloc;
  __syncthreads();
  if (tid < 32) {
    float m = red[tid * 8];
    for (int k = 1; k < 8; ++k) m = fmaxf(m, red[tid * 8 + k]);
    red[tid] = m;
  }
  __syncthreads();
  if (tid == 0) {
    float m = red[0];
    for (int k = 1; k < 32; ++k) m = fmaxf(m, red[k]);
    red[0] = m;
  }
  __syncthreads();
  float gmax = red[0];
  __syncthreads();

  // exp + block sum
  float ev[8];
  float ssum = 0.f;
#pragma unroll
  for (int k = 0; k < 8; ++k) {
    ev[k] = __expf(p_lds[tid + 256 * k] - gmax);
    ssum += ev[k];
  }
  red[tid] = ssum;
  __syncthreads();
  if (tid < 32) {
    float m = 0.f;
    for (int k = 0; k < 8; ++k) m += red[tid * 8 + k];
    red[tid] = m;
  }
  __syncthreads();
  if (tid == 0) {
    float m = 0.f;
    for (int k = 0; k < 32; ++k) m += red[k];
    red[0] = m;
  }
  __syncthreads();
  float inv = 1.f / red[0];
#pragma unroll
  for (int k = 0; k < 8; ++k) {
    int s = tid + 256 * k;
    float p = ev[k] * inv;
    p_lds[s] = p;
    out_scores[(size_t)b * S_ + s] = p;
  }
  __syncthreads();

  // phase B: out[d] = sum_s p[s] * y[s,d]  (thread owns 2 columns, coalesced)
  int d0 = tid * 2;
  float a0 = 0.f, a1 = 0.f;
  for (int s = 0; s < S_; ++s) {
    f32x2 yv = *(const f32x2*)(Yb + (size_t)s * D_ + d0);
    float p = p_lds[s];
    a0 += p * yv.x;
    a1 += p * yv.y;
  }
  out_emb[(size_t)b * D_ + d0] = a0;
  out_emb[(size_t)b * D_ + d0 + 1] = a1;
}

// ---------------------------------------------------------------------------
extern "C" void kernel_launch(void* const* d_in, const int* in_sizes, int n_in,
                              void* d_out, int out_size, void* d_ws,
                              size_t ws_size, hipStream_t stream) {
  const float* x = (const float*)d_in[0];      // [16,2048,512] fp32
  const float* query = (const float*)d_in[1];  // [512,1] fp32
  float* out = (float*)d_out;                  // [16*512] emb ++ [16*2048] scores
  float* Y = (float*)d_ws;                     // y workspace: 64 MB

  size_t smem1 = (size_t)(2 * NK * D_ * 2 + 16384);  // 144 KB
  hipLaunchKernelGGL(attn_fwd_kernel, dim3(B_ * (S_ / MQ)), dim3(THREADS),
                     smem1, stream, x, Y);
  hipLaunchKernelGGL(pool_kernel, dim3(B_), dim3(THREADS), 0, stream, Y, query,
                     out, out + B_ * D_);
}
